// StandardAttentionChunkLayer_7834020348437
// MI455X (gfx1250) — matrix operand order repaired and checked by reading, not compiled, not run on hardware
//
#include <hip/hip_runtime.h>

// MI455X / gfx1250, wave32. All GEMMs via v_wmma_f32_16x16x32_bf16.
// Weights are pre-converted to bf16 once so GEMM inner loops have no
// f32->bf16 VALU chains competing with WMMA issue.

#define DEV __device__ __forceinline__

typedef __attribute__((ext_vector_type(16))) __bf16 v16bf;
typedef __attribute__((ext_vector_type(8)))  float  v8f;

union AFrag { v16bf v; uint4 q[2]; };
union BFrag { v16bf v; uint4 q[2]; unsigned short u[16]; };
union CFrag { v8f  v; float f[8]; };

DEV unsigned short f2bf(float x) {               // round-to-nearest-even f32 -> bf16
  unsigned int u = __float_as_uint(x);
  u += 0x7FFFu + ((u >> 16) & 1u);
  return (unsigned short)(u >> 16);
}

// A-matrix 16x32 bf16 fragment (ISA 7.12.2): lane holds row (lane&15);
// lanes 0-15: K = {k0..k0+7, k0+16..k0+23}; lanes 16-31: +8.
DEV v16bf load_a_frag(const unsigned short* X, size_t row, int k0, int ld) {
  int lane = threadIdx.x & 31;
  const unsigned short* p = X + row * (size_t)ld + k0 + ((lane >> 4) << 3);
  AFrag a;
  a.q[0] = *(const uint4*)(p);
  a.q[1] = *(const uint4*)(p + 16);
  return a.v;
}

// B-matrix 32x16 bf16 fragment: lane holds column col; same K order as A:
// lanes 0-15: K = {k0..k0+7, k0+16..k0+23}; lanes 16-31: +8.
// Bsrc is [N][K] row-major (we dot over its rows), so loads are contiguous.
DEV v16bf load_b_frag_bf16(const unsigned short* Bsrc, size_t col, int k0, int ld) {
  int lane = threadIdx.x & 31;
  const unsigned short* p = Bsrc + col * (size_t)ld + k0 + ((lane >> 4) << 3);
  BFrag b;
  b.q[0] = *(const uint4*)(p);
  b.q[1] = *(const uint4*)(p + 16);
  return b.v;
}

DEV v8f wmma_bf16(v16bf a, v16bf b, v8f c) {
  return __builtin_amdgcn_wmma_f32_16x16x32_bf16(false, a, false, b, (short)0, c,
                                                 false, false);
}

// ---------------------------------------------------------------- prep kernels

__global__ void f32_to_bf16_kernel(const float* __restrict__ x,
                                   unsigned short* __restrict__ y, long long n) {
  long long i = (long long)blockIdx.x * blockDim.x + threadIdx.x;
  if (i < n) y[i] = f2bf(x[i]);
}

// cached_k -> fp32 full_k output rows [0,P) + bf16 kb rows [0,P)
__global__ void prep_k_kernel(const float* __restrict__ src,
                              float* __restrict__ dstF,
                              unsigned short* __restrict__ dstB,
                              int Pfx, int NK, int D, long long total) {
  long long i = (long long)blockIdx.x * blockDim.x + threadIdx.x;
  if (i >= total) return;
  int d = (int)(i % D);
  long long t = i / D;
  int p = (int)(t % Pfx);
  int b = (int)(t / Pfx);
  float v = src[i];
  size_t o = ((size_t)b * NK + p) * D + d;
  dstF[o] = v;
  dstB[o] = f2bf(v);
}

// cached_v -> fp32 full_v output + bf16 vT[b][d][n] (LDS tile transpose)
__global__ void prep_v_kernel(const float* __restrict__ src,
                              float* __restrict__ dstF,
                              unsigned short* __restrict__ vT,
                              int Pfx, int NK, int D) {
  __shared__ unsigned short sh[64][65];
  int b  = blockIdx.z;
  int p0 = blockIdx.x * 64, d0 = blockIdx.y * 64;
  int tx = threadIdx.x & 63, ty = threadIdx.x >> 6;  // 256 threads
  const float* s = src + ((size_t)b * Pfx + p0) * D + d0;
  float* dF = dstF + ((size_t)b * NK + p0) * D + d0;
#pragma unroll
  for (int rr = 0; rr < 64; rr += 4) {
    int p = rr + ty;
    float v = s[(size_t)p * D + tx];
    dF[(size_t)p * D + tx] = v;
    sh[p][tx] = f2bf(v);
  }
  __syncthreads();
  unsigned short* dT = vT + ((size_t)b * D + d0) * NK + p0;
#pragma unroll
  for (int rr = 0; rr < 64; rr += 4) {
    int d = rr + ty;
    dT[(size_t)d * NK + tx] = sh[tx][d];
  }
}

// ----------------------------------------------------- linear: y = x @ W^T + b
// X: bf16 [Mtot][K]; Wb: bf16 [N][K] (pre-converted); per-wave 16(M) x 64(N)
// tile, 4 waves/block. outF: fp32 at b*fBS + (fRowOff+l)*N + n; outB: bf16,
// mode1 row-major, mode2 transposed ([N][rows], packed 16B stores).
__global__ void linear_wmma_kernel(
    const unsigned short* __restrict__ X, const unsigned short* __restrict__ Wb,
    const float* __restrict__ bias,
    float* __restrict__ outF, long long fBS, int fRowOff,
    unsigned short* __restrict__ outB, int bMode, long long bBS, int bLD, int bRowOff,
    int Lrows, int K, int N) {
  int wave = threadIdx.x >> 5;
  int lane = threadIdx.x & 31;
  int m0 = (blockIdx.x * 4 + wave) * 16;
  int n0 = blockIdx.y * 64;
  int b  = m0 / Lrows;
  int l0 = m0 - b * Lrows;
  int hi = lane >> 4;
  size_t rowA = (size_t)m0 + (lane & 15);

  CFrag c[4];
#pragma unroll
  for (int t = 0; t < 4; ++t)
#pragma unroll
    for (int r = 0; r < 8; ++r) c[t].f[r] = 0.0f;

  for (int k0 = 0; k0 < K; k0 += 32) {
    v16bf a = load_a_frag(X, rowA, k0, K);
#pragma unroll
    for (int t = 0; t < 4; ++t) {
      v16bf bm = load_b_frag_bf16(Wb, (size_t)(n0 + t * 16 + (lane & 15)), k0, K);
      c[t].v = wmma_bf16(a, bm, c[t].v);
    }
  }

#pragma unroll
  for (int t = 0; t < 4; ++t) {
    int n = n0 + t * 16 + (lane & 15);
    float bv = bias ? bias[n] : 0.0f;
    float vals[8];
#pragma unroll
    for (int r = 0; r < 8; ++r) vals[r] = c[t].f[r] + bv;
    if (outF) {
#pragma unroll
      for (int r = 0; r < 8; ++r) {
        int l = l0 + hi * 8 + r;
        outF[(size_t)b * fBS + (size_t)(fRowOff + l) * N + n] = vals[r];
      }
    }
    if (bMode == 1) {
#pragma unroll
      for (int r = 0; r < 8; ++r) {
        int l = l0 + hi * 8 + r;
        outB[(size_t)b * bBS + (size_t)(bRowOff + l) * bLD + n] = f2bf(vals[r]);
      }
    } else if (bMode == 2) {  // transposed: 8 consecutive rows -> one 16B store
      union { uint4 q; unsigned short u[8]; } pk;
#pragma unroll
      for (int r = 0; r < 8; ++r) pk.u[r] = f2bf(vals[r]);
      *(uint4*)(outB + (size_t)b * bBS + (size_t)n * bLD + (bRowOff + l0 + hi * 8)) = pk.q;
    }
  }
}

// ------------------------------------------- attention: P = softmax(mask(QK^T))
// Block = 64 query rows (4 waves x 16 rows), streams 64-key tiles.
// Two-pass online softmax; writes normalized probs (bf16) to Pm[b][l][n].
__global__ void attn_softmax_kernel(
    const unsigned short* __restrict__ Qb, const unsigned short* __restrict__ Kb,
    unsigned short* __restrict__ Pm, int Lr, int NK, int D, int Pfx, float scale) {
  int wave = threadIdx.x >> 5;
  int lane = threadIdx.x & 31;
  int qt = blockIdx.x;
  int b  = blockIdx.y;
  int m0 = qt * 64 + wave * 16;
  int hi = lane >> 4;
  const unsigned short* Q  = Qb + (size_t)b * Lr * D;
  const unsigned short* Kc = Kb + (size_t)b * NK * D;
  size_t rowA = (size_t)m0 + (lane & 15);
  int nkt = (Pfx + qt * 64) / 64 + 1;   // key tiles covering causal range

  int qrow[8];
#pragma unroll
  for (int r = 0; r < 8; ++r) qrow[r] = m0 + hi * 8 + r;

  float run_m[8], run_l[8];
#pragma unroll
  for (int r = 0; r < 8; ++r) { run_m[r] = -__builtin_inff(); run_l[r] = 0.0f; }

  // ---- pass 1: row max & sum
  for (int kt = 0; kt < nkt; ++kt) {
    CFrag c[4];
#pragma unroll
    for (int t = 0; t < 4; ++t)
#pragma unroll
      for (int r = 0; r < 8; ++r) c[t].f[r] = 0.0f;
    for (int k0 = 0; k0 < D; k0 += 32) {
      v16bf a = load_a_frag(Q, rowA, k0, D);
#pragma unroll
      for (int t = 0; t < 4; ++t) {
        v16bf bm = load_b_frag_bf16(Kc, (size_t)(kt * 64 + t * 16 + (lane & 15)), k0, D);
        c[t].v = wmma_bf16(a, bm, c[t].v);
      }
    }
    float tmax[8];
#pragma unroll
    for (int r = 0; r < 8; ++r) tmax[r] = -__builtin_inff();
#pragma unroll
    for (int t = 0; t < 4; ++t) {
      int n = kt * 64 + t * 16 + (lane & 15);
#pragma unroll
      for (int r = 0; r < 8; ++r) {
        float v = (n > Pfx + qrow[r]) ? -__builtin_inff() : c[t].f[r] * scale;
        tmax[r] = fmaxf(tmax[r], v);
      }
    }
#pragma unroll
    for (int off = 1; off < 16; off <<= 1)
#pragma unroll
      for (int r = 0; r < 8; ++r) tmax[r] = fmaxf(tmax[r], __shfl_xor(tmax[r], off));
    float nm[8], ts[8];
#pragma unroll
    for (int r = 0; r < 8; ++r) { nm[r] = fmaxf(run_m[r], tmax[r]); ts[r] = 0.0f; }
#pragma unroll
    for (int t = 0; t < 4; ++t) {
      int n = kt * 64 + t * 16 + (lane & 15);
#pragma unroll
      for (int r = 0; r < 8; ++r) {
        float v = (n > Pfx + qrow[r]) ? -__builtin_inff() : c[t].f[r] * scale;
        ts[r] += __expf(v - nm[r]);
      }
    }
#pragma unroll
    for (int off = 1; off < 16; off <<= 1)
#pragma unroll
      for (int r = 0; r < 8; ++r) ts[r] += __shfl_xor(ts[r], off);
#pragma unroll
    for (int r = 0; r < 8; ++r) {
      run_l[r] = run_l[r] * __expf(run_m[r] - nm[r]) + ts[r];
      run_m[r] = nm[r];
    }
  }
  float inv_l[8];
#pragma unroll
  for (int r = 0; r < 8; ++r) inv_l[r] = 1.0f / run_l[r];

  // ---- pass 2: recompute scores, write normalized bf16 probabilities
  unsigned short* Pout = Pm + (size_t)b * Lr * NK;
  for (int kt = 0; kt < nkt; ++kt) {
    CFrag c[4];
#pragma unroll
    for (int t = 0; t < 4; ++t)
#pragma unroll
      for (int r = 0; r < 8; ++r) c[t].f[r] = 0.0f;
    for (int k0 = 0; k0 < D; k0 += 32) {
      v16bf a = load_a_frag(Q, rowA, k0, D);
#pragma unroll
      for (int t = 0; t < 4; ++t) {
        v16bf bm = load_b_frag_bf16(Kc, (size_t)(kt * 64 + t * 16 + (lane & 15)), k0, D);
        c[t].v = wmma_bf16(a, bm, c[t].v);
      }
    }
#pragma unroll
    for (int t = 0; t < 4; ++t) {
      int n = kt * 64 + t * 16 + (lane & 15);
#pragma unroll
      for (int r = 0; r < 8; ++r) {
        float v = (n > Pfx + qrow[r]) ? -__builtin_inff() : c[t].f[r] * scale;
        float p = __expf(v - run_m[r]) * inv_l[r];   // masked -> 0
        Pout[(size_t)qrow[r] * NK + n] = f2bf(p);
      }
    }
  }
}

// --------------------------------------------------- attended = P @ V (bf16^2)
// B-source is vT[b][d][n] (row-major over n) -> contiguous fragment loads.
// K loop bounded by the causally-written range of P (rest is exactly zero).
__global__ void pv_wmma_kernel(
    const unsigned short* __restrict__ Pm, const unsigned short* __restrict__ vT,
    unsigned short* __restrict__ attO, int Lr, int NK, int D, int Pfx) {
  int wave = threadIdx.x >> 5;
  int lane = threadIdx.x & 31;
  int m0 = (blockIdx.x * 4 + wave) * 16;
  int d0 = blockIdx.y * 64;
  int b  = m0 / Lr;
  int l0 = m0 - b * Lr;
  int hi = lane >> 4;
  const unsigned short* A  = Pm + (size_t)b * Lr * NK;
  const unsigned short* Bv = vT + (size_t)b * D * NK;
  size_t rowA = (size_t)l0 + (lane & 15);
  int qt = l0 >> 6;
  int kEnd = Pfx + qt * 64 + 64;
  if (kEnd > NK) kEnd = NK;

  CFrag c[4];
#pragma unroll
  for (int t = 0; t < 4; ++t)
#pragma unroll
    for (int r = 0; r < 8; ++r) c[t].f[r] = 0.0f;

  for (int k0 = 0; k0 < kEnd; k0 += 32) {
    v16bf a = load_a_frag(A, rowA, k0, NK);
#pragma unroll
    for (int t = 0; t < 4; ++t) {
      v16bf bm = load_b_frag_bf16(Bv, (size_t)(d0 + t * 16 + (lane & 15)), k0, NK);
      c[t].v = wmma_bf16(a, bm, c[t].v);
    }
  }
#pragma unroll
  for (int t = 0; t < 4; ++t) {
    int d = d0 + t * 16 + (lane & 15);
#pragma unroll
    for (int r = 0; r < 8; ++r) {
      int l = l0 + hi * 8 + r;
      attO[((size_t)b * Lr + l) * D + d] = f2bf(c[t].f[r]);
    }
  }
}

// -------------------------------------------------------------------- launcher

extern "C" void kernel_launch(void* const* d_in, const int* in_sizes, int n_in,
                              void* d_out, int out_size, void* d_ws, size_t ws_size,
                              hipStream_t stream) {
  (void)in_sizes; (void)n_in; (void)out_size; (void)ws_size;
  const float* chunk   = (const float*)d_in[0];
  const float* cachedK = (const float*)d_in[1];
  const float* cachedV = (const float*)d_in[2];
  const float* Wq = (const float*)d_in[3];
  const float* bq = (const float*)d_in[4];
  const float* Wk = (const float*)d_in[5];
  const float* bk = (const float*)d_in[6];
  const float* Wv = (const float*)d_in[7];
  const float* bv = (const float*)d_in[8];
  const float* Wo = (const float*)d_in[9];
  const float* bo = (const float*)d_in[10];

  const int Bn = 4, L = 2048, Pfx = 2048, D = 1024;
  const int NK = Pfx + L;
  const int Mtot = Bn * L;

  float* outO  = (float*)d_out;                       // [B,L,D]
  float* fullK = outO + (size_t)Bn * L * D;           // [B,NK,D]
  float* fullV = fullK + (size_t)Bn * NK * D;         // [B,NK,D]

  char* w = (char*)d_ws;
  unsigned short* xb = (unsigned short*)w; w += (size_t)Mtot * D * 2;      // chunk bf16
  unsigned short* qb = (unsigned short*)w; w += (size_t)Mtot * D * 2;      // q bf16
  unsigned short* kb = (unsigned short*)w; w += (size_t)Bn * NK * D * 2;   // full_k bf16
  unsigned short* vT = (unsigned short*)w; w += (size_t)Bn * D * NK * 2;   // full_v^T bf16
  unsigned short* Pm = (unsigned short*)w; w += (size_t)Bn * L * NK * 2;   // probs bf16
  unsigned short* wqb = (unsigned short*)w; w += (size_t)D * D * 2;        // Wq bf16
  unsigned short* wkb = (unsigned short*)w; w += (size_t)D * D * 2;        // Wk bf16
  unsigned short* wvb = (unsigned short*)w; w += (size_t)D * D * 2;        // Wv bf16
  unsigned short* wob = (unsigned short*)w; w += (size_t)D * D * 2;        // Wo bf16
  unsigned short* attb = xb;  // reuse: xb dead after QKV projections

  {
    long long n = (long long)Mtot * D;
    f32_to_bf16_kernel<<<dim3((unsigned)((n + 255) / 256)), dim3(256), 0, stream>>>(
        chunk, xb, n);
  }
  {
    long long n = (long long)D * D;
    dim3 g((unsigned)((n + 255) / 256)), t(256);
    f32_to_bf16_kernel<<<g, t, 0, stream>>>(Wq, wqb, n);
    f32_to_bf16_kernel<<<g, t, 0, stream>>>(Wk, wkb, n);
    f32_to_bf16_kernel<<<g, t, 0, stream>>>(Wv, wvb, n);
    f32_to_bf16_kernel<<<g, t, 0, stream>>>(Wo, wob, n);
  }
  {
    long long n = (long long)Bn * Pfx * D;
    prep_k_kernel<<<dim3((unsigned)((n + 255) / 256)), dim3(256), 0, stream>>>(
        cachedK, fullK, kb, Pfx, NK, D, n);
  }
  prep_v_kernel<<<dim3(Pfx / 64, D / 64, Bn), dim3(256), 0, stream>>>(
      cachedV, fullV, vT, Pfx, NK, D);

  dim3 gl(Mtot / 64, D / 64), bl(128);
  // q = chunk @ Wq^T + bq  -> qb (bf16)
  linear_wmma_kernel<<<gl, bl, 0, stream>>>(xb, wqb, bq,
      nullptr, 0, 0,
      qb, 1, (long long)L * D, D, 0,
      L, D, D);
  // new_k -> full_k rows [P,P+L) (fp32) + kb rows [P,P+L) (bf16)
  linear_wmma_kernel<<<gl, bl, 0, stream>>>(xb, wkb, bk,
      fullK, (long long)NK * D, Pfx,
      kb, 1, (long long)NK * D, D, Pfx,
      L, D, D);
  // new_v -> full_v rows [P,P+L) (fp32) + vT cols [P,P+L) (bf16 transposed)
  linear_wmma_kernel<<<gl, bl, 0, stream>>>(xb, wvb, bv,
      fullV, (long long)NK * D, Pfx,
      vT, 2, (long long)D * NK, NK, Pfx,
      L, D, D);

  attn_softmax_kernel<<<dim3(L / 64, Bn), bl, 0, stream>>>(
      qb, kb, Pm, L, NK, D, Pfx, 1.0f / 32.0f);

  pv_wmma_kernel<<<dim3(Mtot / 64, D / 64), bl, 0, stream>>>(
      Pm, vT, attb, L, NK, D, Pfx);

  // out = attended @ Wo^T + bo -> fp32 output
  linear_wmma_kernel<<<gl, bl, 0, stream>>>(attb, wob, bo,
      outO, (long long)L * D, 0,
      nullptr, 0, 0, D, 0,
      L, D, D);
}
